// Tharvexal4Attention_83846351553135
// MI455X (gfx1250) — matrix operand order repaired and checked by reading, not compiled, not run on hardware
//
#include <hip/hip_runtime.h>
#include <hip/hip_bf16.h>

// ---------------------------------------------------------------------------
// GQA attention block for MI455X (gfx1250, wave32, WMMA + TDM + async-LDS).
// Pipeline: cvt->bf16, fused QKV GEMM (TDM double-buffered WMMA), RoPE,
// flash attention (async-LDS staged, WMMA), output projection GEMM.
// ---------------------------------------------------------------------------

typedef __attribute__((ext_vector_type(16))) __bf16 v16bf;
typedef __attribute__((ext_vector_type(8)))  float  v8f;
typedef unsigned int u32x4 __attribute__((ext_vector_type(4)));
typedef int          i32x8 __attribute__((ext_vector_type(8)));
typedef int          i32x4 __attribute__((ext_vector_type(4)));

#define DEV __device__ __forceinline__

static constexpr int B_   = 2;
static constexpr int S_   = 2048;
static constexpr int D_   = 4096;
static constexpr int NH_  = 32;
static constexpr int NKV_ = 8;
static constexpr int HD_  = 128;
static constexpr int NQKV_ = D_ + 2 * NKV_ * HD_;   // 6144

DEV unsigned short f2bf(float f) {
  unsigned u = __float_as_uint(f);
  u += 0x7FFFu + ((u >> 16) & 1u);           // round-to-nearest-even
  return (unsigned short)(u >> 16);
}
DEV float bf2f(unsigned short h) { return __uint_as_float(((unsigned)h) << 16); }

// Build a v16bf WMMA fragment from two 16-byte chunks.
DEV v16bf load_frag(const unsigned short* p0, const unsigned short* p1) {
  union { uint4 u[2]; v16bf v; } f;
  f.u[0] = *reinterpret_cast<const uint4*>(p0);
  f.u[1] = *reinterpret_cast<const uint4*>(p1);
  return f.v;
}

// Per-lane async global->LDS copy, tracked on ASYNCcnt.
DEV void async_ld_b128(unsigned lds_off, unsigned long long gaddr) {
  asm volatile("global_load_async_to_lds_b128 %0, %1, off"
               :: "v"(lds_off), "v"(gaddr) : "memory");
}
DEV void wait_asynccnt0() {
  asm volatile("s_wait_asynccnt 0" ::: "memory");
}

// ---------------------------------------------------------------------------
// TDM: load a 128-row x 32-col bf16 tile of a [rows_total, 4096] bf16 tensor
// into LDS at lds_off, inserting 32B of LDS padding after every 64B row
// (=> padded LDS row stride 96B = SROW elems), via one D# descriptor.
//   group0: count=1 | lds_addr | global_addr | type=2
//   group1: data_size=1(2B), pad_enable, pad_interval=3(16 dwords),
//           pad_amount=7(8 dwords), tensor_dim0=4096, tensor_dim1=rows_total,
//           tile_dim0=32, tile_dim1=128, tensor_dim0_stride=4096
// ---------------------------------------------------------------------------
DEV void tdm_load_tile(unsigned lds_off, const unsigned short* gptr, int rows_total) {
  unsigned long long ga = (unsigned long long)(uintptr_t)gptr;
  u32x4 g0;
  g0[0] = 1u;                                             // count = 1
  g0[1] = lds_off;                                        // lds_addr
  g0[2] = (unsigned)ga;                                   // global_addr[31:0]
  g0[3] = (unsigned)((ga >> 32) & 0x01FFFFFFu) | 0x80000000u;  // addr[56:32]|type=2
  i32x8 g1;
  g1[0] = 0x0ED10000;                                     // dsz=2B,pad on,int=16dw,amt=8dw
  g1[1] = (int)((4096u & 0xFFFFu) << 16);                 // tensor_dim0 lo16
  g1[2] = (int)(((unsigned)rows_total & 0xFFFFu) << 16);  // dim0 hi=0 | dim1 lo16
  g1[3] = (int)((((unsigned)rows_total >> 16) & 0xFFFFu) | (32u << 16)); // dim1 hi | tile_dim0
  g1[4] = 128;                                            // tile_dim1 | tile_dim2=0
  g1[5] = 4096;                                           // tensor_dim0_stride lo32
  g1[6] = 0;                                              // stride hi | dim1_stride lo
  g1[7] = 0;
  i32x4 z4 = {0, 0, 0, 0};
  i32x8 z8 = {0, 0, 0, 0, 0, 0, 0, 0};
  // 6-arg form (clang-23 / therock-10.0 headers).
  __builtin_amdgcn_tensor_load_to_lds(g0, g1, z4, z4, z8, 0);
}

// ---------------------------------------------------------------------------
// Kernel 1: f32 -> bf16 convert (grid-stride)
// ---------------------------------------------------------------------------
__global__ void cvt_f32_bf16(const float* __restrict__ src,
                             unsigned short* __restrict__ dst, size_t n) {
  size_t stride = (size_t)gridDim.x * blockDim.x;
  for (size_t i = (size_t)blockIdx.x * blockDim.x + threadIdx.x; i < n; i += stride)
    dst[i] = f2bf(src[i]);
}

// ---------------------------------------------------------------------------
// Kernel 2/5: bf16 GEMM, C = A[M,K] * Bw[N,K]^T, f32 accumulate via WMMA.
//  Block tile 128x128, K-tile 32, double-buffered LDS fed by TDM.
//  256 threads = 8 waves, wave tile 32x64 (8 WMMAs per k-step).
//  MODE 0: write f32 to C (row-major [M,N])
//  MODE 1: QKV epilogue: bf16 scatter into Q[B,NH,S,HD], K/V[B,NKV,S,HD]
// ---------------------------------------------------------------------------
static constexpr int SROW = 48;   // padded LDS row stride (elems); 96B

template <int MODE>
__global__ __launch_bounds__(256)
void gemm_bf16(const unsigned short* __restrict__ A,
               const unsigned short* __restrict__ Bw,
               float* __restrict__ C, int M, int N, int K,
               unsigned short* __restrict__ Qo,
               unsigned short* __restrict__ Ko,
               unsigned short* __restrict__ Vo) {
  __shared__ __attribute__((aligned(16))) unsigned short sA[2][128 * SROW];
  __shared__ __attribute__((aligned(16))) unsigned short sB[2][128 * SROW];

  const int tid  = threadIdx.x;
  const int m0b  = blockIdx.y * 128;
  const int n0b  = blockIdx.x * 128;
  const int wave = tid >> 5, lane = tid & 31;
  const int wm = wave >> 1, wn = wave & 1;
  const int m0 = wm * 32, n0 = wn * 64;
  const int lm = lane & 15, half = lane >> 4;

  v8f acc[2][4] = {};
  const int KT = K >> 5;

  // Prologue: TDM stage tile 0 (one wave issues; DMA engine does the copy).
  if (wave == 0) {
    tdm_load_tile((unsigned)(uintptr_t)&sA[0][0], A + (size_t)m0b * K, M);
    tdm_load_tile((unsigned)(uintptr_t)&sB[0][0], Bw + (size_t)n0b * K, N);
    __builtin_amdgcn_s_wait_tensorcnt(0);
  }
  __syncthreads();

#pragma unroll 1
  for (int kt = 0; kt < KT; kt++) {
    // Prefetch next K-tile into the other buffer while we compute.
    if (wave == 0 && kt + 1 < KT) {
      int k0n = (kt + 1) << 5;
      tdm_load_tile((unsigned)(uintptr_t)&sA[(kt + 1) & 1][0],
                    A + (size_t)m0b * K + k0n, M);
      tdm_load_tile((unsigned)(uintptr_t)&sB[(kt + 1) & 1][0],
                    Bw + (size_t)n0b * K + k0n, N);
    }
    const unsigned short* cA = sA[kt & 1];
    const unsigned short* cB = sB[kt & 1];

    v16bf aF[2], bF[4];
#pragma unroll
    for (int t = 0; t < 2; t++) {
      const unsigned short* base = &cA[(m0 + t * 16 + lm) * SROW];
      aF[t] = load_frag(base + half * 8, base + 16 + half * 8);
    }
#pragma unroll
    for (int nt = 0; nt < 4; nt++) {
      const unsigned short* base = &cB[(n0 + nt * 16 + lm) * SROW + half * 8];
      bF[nt] = load_frag(base, base + 16);
    }
#pragma unroll
    for (int t = 0; t < 2; t++)
#pragma unroll
      for (int nt = 0; nt < 4; nt++)
        acc[t][nt] = __builtin_amdgcn_wmma_f32_16x16x32_bf16(
            false, aF[t], false, bF[nt], (short)0, acc[t][nt], false, false);

    if (wave == 0) __builtin_amdgcn_s_wait_tensorcnt(0);
    __syncthreads();
  }

  // Epilogue. C layout: lane = N (within 16), VGPR r -> M = r + 8*(lane/16).
#pragma unroll
  for (int t = 0; t < 2; t++)
#pragma unroll
    for (int nt = 0; nt < 4; nt++)
#pragma unroll
      for (int r = 0; r < 8; r++) {
        int m = m0b + m0 + t * 16 + r + 8 * half;
        int n = n0b + n0 + nt * 16 + lm;
        float v = acc[t][nt][r];
        if (MODE == 0) {
          C[(size_t)m * N + n] = v;
        } else {
          int b = m >> 11, s = m & (S_ - 1);
          unsigned short bv = f2bf(v);
          if (n < D_) {
            int hh = n >> 7, d = n & 127;
            Qo[((((size_t)b * NH_ + hh) * S_ + s) << 7) + d] = bv;
          } else if (n < D_ + NKV_ * HD_) {
            int nn = n - D_; int hh = nn >> 7, d = nn & 127;
            Ko[((((size_t)b * NKV_ + hh) * S_ + s) << 7) + d] = bv;
          } else {
            int nn = n - (D_ + NKV_ * HD_); int hh = nn >> 7, d = nn & 127;
            Vo[((((size_t)b * NKV_ + hh) * S_ + s) << 7) + d] = bv;
          }
        }
      }
}

// ---------------------------------------------------------------------------
// Kernel 3: RoPE in-place on bf16 [rows, 128] (rows = B*H*S, pos = row % S)
// ---------------------------------------------------------------------------
__global__ void rope_kernel(unsigned short* __restrict__ Xh, size_t rows) {
  size_t total = rows * 64;
  size_t stride = (size_t)gridDim.x * blockDim.x;
  for (size_t i = (size_t)blockIdx.x * blockDim.x + threadIdx.x; i < total; i += stride) {
    int    j   = (int)(i & 63);
    size_t row = i >> 6;
    int    s   = (int)(row & (S_ - 1));
    float inv = __powf(10000.0f, -(float)j * (1.0f / 64.0f));
    float ang = (float)s * inv;
    float cs = __cosf(ang), sn = __sinf(ang);
    unsigned short* p = Xh + row * HD_;
    float x1 = bf2f(p[j]), x2 = bf2f(p[j + 64]);
    p[j]      = f2bf(x1 * cs - x2 * sn);
    p[j + 64] = f2bf(x2 * cs + x1 * sn);
  }
}

// ---------------------------------------------------------------------------
// Kernel 4: flash attention, causal, GQA (4 Q-heads per KV head).
//  grid = (S/128, B*NH); 256 threads = 8 waves; each wave owns 16 Q rows.
//  K tile async-staged to LDS (ASYNCcnt), V tile transposed through VGPRs;
//  online softmax in f32; 32 WMMAs per wave per 64-key block.
// ---------------------------------------------------------------------------
static constexpr int KROW = 136;  // K-tile LDS stride (elems), 272B
static constexpr int VROW = 72;   // V^T / P LDS stride (elems), 144B

__global__ __launch_bounds__(256)
void flash_attn(const unsigned short* __restrict__ Q,
                const unsigned short* __restrict__ Kb,
                const unsigned short* __restrict__ Vb,
                unsigned short* __restrict__ O) {
  __shared__ __attribute__((aligned(16))) unsigned short sK[64 * KROW];
  __shared__ __attribute__((aligned(16))) unsigned short sVT[128 * VROW];
  __shared__ __attribute__((aligned(16))) unsigned short sP[8 * 16 * VROW];

  const int tid  = threadIdx.x;
  const int qblk = blockIdx.x;             // 0..15
  const int bh   = blockIdx.y;             // 0..63
  const int b = bh >> 5, h = bh & 31, kvh = h >> 2;
  const int wave = tid >> 5, lane = tid & 31;
  const int lm = lane & 15, half = lane >> 4;
  const int q0 = qblk * 128;
  const int qw = q0 + wave * 16;           // this wave's first query row

  // Q fragments for this wave's 16x128 tile (A-matrix layout), loaded once.
  v16bf qF[4];
  {
    const unsigned short* qrow =
        Q + (((size_t)b * NH_ + h) * S_ + qw + lm) * HD_;
#pragma unroll
    for (int kf = 0; kf < 4; kf++)
      qF[kf] = load_frag(qrow + kf * 32 + half * 8,
                         qrow + kf * 32 + 16 + half * 8);
  }

  v8f acc[8] = {};
  float mrow[8], lrow[8];
#pragma unroll
  for (int r = 0; r < 8; r++) { mrow[r] = -3.0e38f; lrow[r] = 0.0f; }
  const float scale = 0.08838834764831845f;  // 1/sqrt(128)

  const unsigned short* Kbase = Kb + ((size_t)b * NKV_ + kvh) * S_ * HD_;
  const unsigned short* Vbase = Vb + ((size_t)b * NKV_ + kvh) * S_ * HD_;
  unsigned short* sPw = sP + wave * 16 * VROW;

  const int ktmax = qblk * 2 + 2;          // causal early termination
#pragma unroll 1
  for (int kt = 0; kt < ktmax; kt++) {
    // Stage K tile via async global->LDS; V tile transposed through VGPRs.
#pragma unroll
    for (int i = 0; i < 4; i++) {
      int chunk = tid + i * 256;           // 0..1023
      int krow  = chunk >> 4;              // key index 0..63
      int c8    = (chunk & 15) * 8;        // d offset
      size_t g  = (size_t)(kt * 64 + krow) * HD_ + c8;
      async_ld_b128((unsigned)(uintptr_t)&sK[krow * KROW + c8],
                    (unsigned long long)(uintptr_t)&Kbase[g]);
      union { uint4 u; unsigned short s[8]; } vv;
      vv.u = *(const uint4*)&Vbase[g];
#pragma unroll
      for (int j = 0; j < 8; j++) sVT[(c8 + j) * VROW + krow] = vv.s[j];
    }
    wait_asynccnt0();
    __syncthreads();

    // Scores S = Q K^T : 4 col-tiles x 4 k-steps = 16 WMMAs.
    v8f sc[4] = {};
#pragma unroll
    for (int nt = 0; nt < 4; nt++) {
      const unsigned short* kr = &sK[(nt * 16 + lm) * KROW];
#pragma unroll
      for (int kf = 0; kf < 4; kf++) {
        v16bf bK = load_frag(kr + kf * 32 + half * 8,
                             kr + kf * 32 + half * 8 + 16);
        sc[nt] = __builtin_amdgcn_wmma_f32_16x16x32_bf16(
            false, qF[kf], false, bK, (short)0, sc[nt], false, false);
      }
    }

    // Online softmax (rows live across 16-lane halves of the C layout).
    float pv[4][8];
#pragma unroll
    for (int r = 0; r < 8; r++) {
      int qpos = qw + r + 8 * half;
      float mx = -3.0e38f;
#pragma unroll
      for (int nt = 0; nt < 4; nt++) {
        int kpos = kt * 64 + nt * 16 + lm;
        float x = sc[nt][r] * scale;
        if (kpos > qpos) x = -3.0e38f;     // causal mask
        sc[nt][r] = x;
        mx = fmaxf(mx, x);
      }
#pragma unroll
      for (int off = 1; off < 16; off <<= 1)
        mx = fmaxf(mx, __shfl_xor(mx, off, 16));
      float mnew  = fmaxf(mrow[r], mx);
      float alpha = __expf(mrow[r] - mnew);
      mrow[r] = mnew;
      float rs = 0.0f;
#pragma unroll
      for (int nt = 0; nt < 4; nt++) {
        float p = __expf(sc[nt][r] - mnew);
        pv[nt][r] = p;
        rs += p;
      }
#pragma unroll
      for (int off = 1; off < 16; off <<= 1)
        rs += __shfl_xor(rs, off, 16);
      lrow[r] = lrow[r] * alpha + rs;
#pragma unroll
      for (int dt = 0; dt < 8; dt++) acc[dt][r] *= alpha;
    }

    // Transpose P (C layout -> A layout) through this wave's LDS region.
#pragma unroll
    for (int r = 0; r < 8; r++) {
      int m = r + 8 * half;
#pragma unroll
      for (int nt = 0; nt < 4; nt++)
        sPw[m * VROW + nt * 16 + lm] = f2bf(pv[nt][r]);
    }
    __syncthreads();

    // O += P V : 2 k-steps x 8 d-tiles = 16 WMMAs.
#pragma unroll
    for (int kf2 = 0; kf2 < 2; kf2++) {
      const unsigned short* pr = &sPw[lm * VROW + kf2 * 32];
      v16bf pF = load_frag(pr + half * 8, pr + 16 + half * 8);
#pragma unroll
      for (int dt = 0; dt < 8; dt++) {
        const unsigned short* vr = &sVT[(dt * 16 + lm) * VROW + kf2 * 32 + half * 8];
        v16bf bV = load_frag(vr, vr + 16);
        acc[dt] = __builtin_amdgcn_wmma_f32_16x16x32_bf16(
            false, pF, false, bV, (short)0, acc[dt], false, false);
      }
    }
    __syncthreads();
  }

  // Normalize and write bf16 context into AO[B*S, D] for the Wo GEMM.
#pragma unroll
  for (int r = 0; r < 8; r++) {
    int m = qw + r + 8 * half;
    float inv = 1.0f / lrow[r];
    size_t rowbase = ((size_t)b * S_ + m) * D_ + (size_t)h * HD_;
#pragma unroll
    for (int dt = 0; dt < 8; dt++)
      O[rowbase + dt * 16 + lm] = f2bf(acc[dt][r] * inv);
  }
}

// ---------------------------------------------------------------------------
// Launch
// ---------------------------------------------------------------------------
extern "C" void kernel_launch(void* const* d_in, const int* in_sizes, int n_in,
                              void* d_out, int out_size, void* d_ws, size_t ws_size,
                              hipStream_t stream) {
  const float* X  = (const float*)d_in[0];
  // d_in[1] = attention_mask (causal; computed inline), d_in[2] = position_ids (== arange)
  const float* Wq = (const float*)d_in[3];
  const float* Wk = (const float*)d_in[4];
  const float* Wv = (const float*)d_in[5];
  const float* Wo = (const float*)d_in[6];
  float* out = (float*)d_out;

  const size_t nX   = (size_t)B_ * S_ * D_;          // 16M
  const size_t nWq  = (size_t)D_ * D_;               // 16M
  const size_t nWkv = (size_t)NKV_ * HD_ * D_;       // 4M
  const size_t nQ   = (size_t)B_ * NH_ * S_ * HD_;   // 16M
  const size_t nKV  = (size_t)B_ * NKV_ * S_ * HD_;  // 4M

  unsigned short* Xbf  = (unsigned short*)d_ws;
  unsigned short* Wqkv = Xbf + nX;                   // [6144, 4096] packed
  unsigned short* Wobf = Wqkv + (size_t)NQKV_ * D_;
  unsigned short* Qb   = Wobf + nWq;
  unsigned short* Kbf  = Qb + nQ;
  unsigned short* Vbf  = Kbf + nKV;
  unsigned short* AOb  = Vbf + nKV;                  // [B*S, D] bf16 context

  cvt_f32_bf16<<<8192, 256, 0, stream>>>(X,  Xbf, nX);
  cvt_f32_bf16<<<8192, 256, 0, stream>>>(Wq, Wqkv, nWq);
  cvt_f32_bf16<<<2048, 256, 0, stream>>>(Wk, Wqkv + nWq, nWkv);
  cvt_f32_bf16<<<2048, 256, 0, stream>>>(Wv, Wqkv + nWq + nWkv, nWkv);
  cvt_f32_bf16<<<8192, 256, 0, stream>>>(Wo, Wobf, nWq);

  dim3 gQKV(NQKV_ / 128, (B_ * S_) / 128);           // (48, 32)
  gemm_bf16<1><<<gQKV, 256, 0, stream>>>(Xbf, Wqkv, nullptr,
                                         B_ * S_, NQKV_, D_, Qb, Kbf, Vbf);

  rope_kernel<<<8192, 256, 0, stream>>>(Qb,  (size_t)B_ * NH_  * S_);
  rope_kernel<<<2048, 256, 0, stream>>>(Kbf, (size_t)B_ * NKV_ * S_);

  dim3 gA(S_ / 128, B_ * NH_);                       // (16, 64)
  flash_attn<<<gA, 256, 0, stream>>>(Qb, Kbf, Vbf, AOb);

  dim3 gO(D_ / 128, (B_ * S_) / 128);                // (32, 32)
  gemm_bf16<0><<<gO, 256, 0, stream>>>(AOb, Wobf, out,
                                       B_ * S_, D_, D_,
                                       nullptr, nullptr, nullptr);
}
